// GCN_35828617183475
// MI455X (gfx1250) — matrix operand order repaired and checked by reading, not compiled, not run on hardware
//
#include <hip/hip_runtime.h>
#include <hip/hip_bf16.h>
#include <stdint.h>

// GCN layer: agg = (adj @ x)/deg ; out = row_normalize(sigmoid([x,agg] @ W + b))
// B=8, N=2048, D=128, OUT=128. f16 WMMA (adj 0/1 exact in f16), f32 accumulate.

#define BATCH 8
#define NN    2048
#define DD    128
#define OUTC  128

typedef __attribute__((ext_vector_type(16))) _Float16 v16h;
typedef __attribute__((ext_vector_type(8)))  _Float16 v8h;
typedef __attribute__((ext_vector_type(8)))  float    v8f;

union AFrag { unsigned int w[8]; v16h h; };

// ---------------------------------------------------------------------------
// Prep: input f32 [B][N][D] -> f16 row-major (inR) and f16 transposed [B][D][N]
// (inT).  inT makes every WMMA B-fragment a contiguous 32B load per lane.
// ---------------------------------------------------------------------------
__global__ __launch_bounds__(256) void prep_input_kernel(
    const float* __restrict__ in, _Float16* __restrict__ inR,
    _Float16* __restrict__ inT) {
  __shared__ _Float16 tile[32][33];
  const int nb = blockIdx.x * 32;      // n tile
  const int db = blockIdx.y * 32;      // d tile
  const int b  = blockIdx.z;
  const int tx = threadIdx.x & 31;
  const int ty = threadIdx.x >> 5;
  for (int r = ty; r < 32; r += 8) {
    size_t idx = ((size_t)b * NN + nb + r) * DD + db + tx;
    _Float16 h = (_Float16)in[idx];
    inR[idx] = h;
    tile[r][tx] = h;
  }
  __syncthreads();
  for (int r = ty; r < 32; r += 8) {
    // inT[b][d][n] = in[b][n][d]
    inT[((size_t)b * DD + db + r) * NN + nb + tx] = tile[tx][r];
  }
}

// W f32 [2D][OUT] -> Wt f16 [OUT][2D] (transposed, contiguous K per column)
__global__ __launch_bounds__(256) void prep_w_kernel(
    const float* __restrict__ W, _Float16* __restrict__ Wt) {
  for (int i = threadIdx.x; i < 2 * DD * OUTC; i += 256) {
    int k = i / OUTC, n = i % OUTC;
    Wt[(size_t)n * (2 * DD) + k] = (_Float16)W[i];
  }
}

// Pack two {0,1} ints into one dword of two f16 values, in 2 VALU ops:
// (a | b<<16) * 0x3C00 == (a?1.0h:0) | (b?1.0h:0)<<16   (no cross-half carry)
__device__ __forceinline__ unsigned int pack2(int a, int b,
                                              unsigned int& sAcc) {
  unsigned int p = (unsigned int)a | ((unsigned int)b << 16);
  sAcc += p;  // packed degree accumulation; halves never carry (<= 512)
  return p * 0x3C00u;
}

// ---------------------------------------------------------------------------
// Fused main kernel. 128 threads = 4 waves; wave owns 16 rows x 128 cols.
//
// WMMA f16 register layouts used (A and B share one contiguous k order):
//   A 16x32:  lane L (m=L&15, g=L>>4), half h -> K = h + 16*g
//             => one contiguous 16-half chunk at K = 16g (row-major source).
//   B 32x16:  lane L (n=L&15, g=L>>4), half h -> K = h + 16*g
//             => one contiguous 16-half chunk at K = 16g (column-major source).
//   C 16x16:  lane L, vgpr j -> row M = 8*g + j, col = L&15.
// ---------------------------------------------------------------------------
__global__ __launch_bounds__(128) void gcn_main_kernel(
    const int* __restrict__ adj, const _Float16* __restrict__ inR,
    const _Float16* __restrict__ inT, const _Float16* __restrict__ Wt,
    const float* __restrict__ bias, float* __restrict__ out) {
  __shared__ __align__(32) _Float16 aggLds[4][16 * DD];  // per-wave 16x128 agg

  const int lane = threadIdx.x & 31;
  const int wave = threadIdx.x >> 5;
  const int m = lane & 15;   // A row / B-C column within tile
  const int g = lane >> 4;   // lane group
  const int b = blockIdx.y;
  const int rowBase = blockIdx.x * 64 + wave * 16;

  const int*      arow = adj + ((size_t)b * NN + rowBase + m) * NN;
  const _Float16* bt   = inT + (size_t)b * DD * NN;  // [d][n]

  // ---------------- GEMM1: agg = adj @ x, K = 2048 ----------------
  v8f acc[8] = {};
  unsigned int sAcc = 0;  // packed degree accumulator (lo16 + hi16 halves)
  for (int kb = 0; kb < NN; kb += 32) {
    __builtin_prefetch(arow + kb + 1024, 0, 1);  // stream adj ahead
    int4 q0 = *(const int4*)(arow + kb + 16 * g);
    int4 q1 = *(const int4*)(arow + kb + 16 * g + 4);
    int4 q2 = *(const int4*)(arow + kb + 16 * g + 8);
    int4 q3 = *(const int4*)(arow + kb + 16 * g + 12);
    AFrag A;
    A.w[0] = pack2(q0.x, q0.y, sAcc);
    A.w[1] = pack2(q0.z, q0.w, sAcc);
    A.w[2] = pack2(q1.x, q1.y, sAcc);
    A.w[3] = pack2(q1.z, q1.w, sAcc);
    A.w[4] = pack2(q2.x, q2.y, sAcc);
    A.w[5] = pack2(q2.z, q2.w, sAcc);
    A.w[6] = pack2(q3.x, q3.y, sAcc);
    A.w[7] = pack2(q3.z, q3.w, sAcc);
#pragma unroll
    for (int t = 0; t < 8; ++t) {
      v16h bv = *(const v16h*)(bt + (size_t)(t * 16 + m) * NN + kb + 16 * g);
      acc[t] = __builtin_amdgcn_wmma_f32_16x16x32_f16(
          false, A.h, false, bv, (short)0, acc[t], false, false);
    }
  }
  int deg = (int)(sAcc & 0xFFFFu) + (int)(sAcc >> 16);

  // Degree: lanes L and L+16 hold complementary K halves for row (L&15).
  float degf = (float)deg;
  degf += __shfl_xor(degf, 16, 32);
  if (degf < 1.0f) degf = 1.0f;  // deg==0 -> 1

  // agg /= deg, cast f16, park in per-wave LDS (row-major 16x128).
  _Float16* al = aggLds[wave];
#pragma unroll
  for (int j = 0; j < 8; ++j) {
    float inv = 1.0f / __shfl(degf, 8 * g + j, 32);
#pragma unroll
    for (int t = 0; t < 8; ++t)
      al[(8 * g + j) * DD + t * 16 + m] = (_Float16)(acc[t][j] * inv);
  }

  // ---------------- GEMM2: [x, agg] @ W, K = 256 ----------------
  v8f acc2[8] = {};
  const _Float16* frow = inR + ((size_t)b * NN + rowBase + m) * DD;
#pragma unroll
  for (int kk = 0; kk < 8; ++kk) {
    const int kb = kk * 32;
    v8h lo, hi;
    if (kb < DD) {  // feature half from row-major f16 input
      lo = *(const v8h*)(frow + kb + 16 * g);
      hi = *(const v8h*)(frow + kb + 16 * g + 8);
    } else {        // aggregation half from this wave's LDS tile
      lo = *(const v8h*)(al + m * DD + (kb - DD) + 16 * g);
      hi = *(const v8h*)(al + m * DD + (kb - DD) + 16 * g + 8);
    }
    v16h a2 = __builtin_shufflevector(lo, hi, 0, 1, 2, 3, 4, 5, 6, 7, 8, 9, 10,
                                      11, 12, 13, 14, 15);
#pragma unroll
    for (int t = 0; t < 8; ++t) {
      v16h wv = *(const v16h*)(Wt + (size_t)(t * 16 + m) * (2 * DD) + kb + 16 * g);
      acc2[t] = __builtin_amdgcn_wmma_f32_16x16x32_f16(
          false, a2, false, wv, (short)0, acc2[t], false, false);
    }
  }

  // ---------------- bias + sigmoid + row L2-normalize + store ----------------
  float sumsq[8] = {};
#pragma unroll
  for (int t = 0; t < 8; ++t) {
    float bv = bias[t * 16 + m];
#pragma unroll
    for (int j = 0; j < 8; ++j) {
      float s = 1.0f / (1.0f + __expf(-(acc2[t][j] + bv)));
      acc2[t][j] = s;
      sumsq[j] += s * s;
    }
  }
  // Row M=8g+j spans the 16 lanes of this lane-group: reduce within group.
#pragma unroll
  for (int j = 0; j < 8; ++j) {
#pragma unroll
    for (int o = 1; o < 16; o <<= 1) sumsq[j] += __shfl_xor(sumsq[j], o, 32);
  }
  float* orow = out + ((size_t)b * NN + rowBase) * OUTC;
#pragma unroll
  for (int j = 0; j < 8; ++j) {
    float inv = rsqrtf(sumsq[j]);
#pragma unroll
    for (int t = 0; t < 8; ++t)
      orow[(size_t)(8 * g + j) * OUTC + t * 16 + m] = acc2[t][j] * inv;
  }
}

// ---------------------------------------------------------------------------
extern "C" void kernel_launch(void* const* d_in, const int* in_sizes, int n_in,
                              void* d_out, int out_size, void* d_ws,
                              size_t ws_size, hipStream_t stream) {
  const float* input = (const float*)d_in[0];  // [8,2048,128] f32
  const int*   adj   = (const int*)d_in[1];    // [8,2048,2048] i32
  const float* W     = (const float*)d_in[2];  // [256,128] f32
  const float* bias  = (const float*)d_in[3];  // [128] f32
  float*       out   = (float*)d_out;          // [8,2048,128] f32

  char* ws = (char*)d_ws;
  const size_t inBytes = (size_t)BATCH * NN * DD * sizeof(_Float16);  // 4 MB
  _Float16* inT = (_Float16*)(ws);                 // [B][D][N] f16
  _Float16* inR = (_Float16*)(ws + inBytes);       // [B][N][D] f16
  _Float16* Wt  = (_Float16*)(ws + 2 * inBytes);   // [OUT][2D] f16 (64 KB)

  prep_input_kernel<<<dim3(NN / 32, DD / 32, BATCH), 256, 0, stream>>>(
      input, inR, inT);
  prep_w_kernel<<<1, 256, 0, stream>>>(W, Wt);
  gcn_main_kernel<<<dim3(NN / 64, BATCH), 128, 0, stream>>>(
      adj, inR, inT, Wt, bias, out);
}
